// VLAttHierLstm_31963146616844
// MI455X (gfx1250) — compile-verified, matrix operand order checked by reading, not run on hardware
//
#include <hip/hip_runtime.h>
#include <stdint.h>

// ---------------------------------------------------------------- constants
#define SEQ  4096
#define HID  1024
#define INPD 1024
#define G4   4096      // 4*HID
#define UPL  64        // min(80, floor(sqrt(4096))) = 64; idx[i] = 63 + 64*i
#define NWG_SCAN 16
#define TPB_SCAN 64    // units per scan workgroup (16*64 = 1024 units)

typedef __bf16 bf16_t;
typedef __attribute__((ext_vector_type(16))) __bf16    v16bf;
typedef __attribute__((ext_vector_type(8)))  float     v8f;
typedef __attribute__((ext_vector_type(4)))  uint32_t  u32x4v;
typedef __attribute__((ext_vector_type(4)))  int       i32x4v;
typedef __attribute__((ext_vector_type(8)))  int       i32x8v;

#if defined(__has_builtin)
#  if __has_builtin(__builtin_amdgcn_tensor_load_to_lds)
#    define HAVE_TDM 1
#  endif
#endif
#ifndef HAVE_TDM
#  define HAVE_TDM 0
#endif

// ------------------------------------------------------------- tiny helpers
__device__ __forceinline__ float sigm(float x) { return 1.f / (1.f + __expf(-x)); }

// two packed bf16 -> two f32 (bf16 is the high half of an f32)
__device__ __forceinline__ float2 bf2f(uint32_t u) {
  float2 r;
  r.x = __uint_as_float(u << 16);
  r.y = __uint_as_float(u & 0xffff0000u);
  return r;
}

// Grid-wide rendezvous for the persistent scan kernel. Cluster barrier gives a
// fast SE-local sync if dispatched as a cluster (s_barrier_signal/wait -3,
// S_NOP when ClusterID==0); the generation counter is the portable guarantee.
__device__ __forceinline__ void grid_sync(int* cnt, int* gen, int nwg) {
  __syncthreads();
  if (threadIdx.x == 0) {
    __builtin_amdgcn_s_cluster_barrier();
    __threadfence();
    int g = __hip_atomic_load(gen, __ATOMIC_RELAXED, __HIP_MEMORY_SCOPE_AGENT);
    if (__hip_atomic_fetch_add(cnt, 1, __ATOMIC_ACQ_REL, __HIP_MEMORY_SCOPE_AGENT) == nwg - 1) {
      __hip_atomic_store(cnt, 0, __ATOMIC_RELAXED, __HIP_MEMORY_SCOPE_AGENT);
      __hip_atomic_store(gen, g + 1, __ATOMIC_RELEASE, __HIP_MEMORY_SCOPE_AGENT);
    } else {
      while (__hip_atomic_load(gen, __ATOMIC_ACQUIRE, __HIP_MEMORY_SCOPE_AGENT) == g) {
        __builtin_amdgcn_s_sleep(1);
      }
    }
  }
  __syncthreads();
}

// --------------------------------------------------------------- f32 -> bf16
__global__ void k_cvt_bf16(bf16_t* __restrict__ dst, const float* __restrict__ src, int n) {
  int i = blockIdx.x * blockDim.x + threadIdx.x;
  if (i < n) dst[i] = (bf16_t)src[i];
}

// ------------------------------------------------------------- TDM tile DMA
#if HAVE_TDM
// Generic LDS pointer -> byte offset within the workgroup LDS allocation
// (low 32 bits of the shared-aperture generic address).
__device__ __forceinline__ uint32_t lds_off(const void* p) {
  return (uint32_t)(uintptr_t)p;
}

// Async 2D tile load, bf16 elements: rows x 32 elems, row stride = strideElems.
// D# layout per CDNA5 ISA. Issue once per wave
// (TDM ignores EXEC); tracked on TENSORcnt.
__device__ __forceinline__ void tdm_load_2d(uint32_t ldsAddr, const void* gptr,
                                            uint32_t rows, uint32_t tensorW,
                                            uint32_t tensorH, uint32_t strideElems) {
  uint64_t ga = (uint64_t)(uintptr_t)gptr;
  u32x4v g0;
  g0[0] = 1u;                                            // count=1 (valid user D#)
  g0[1] = ldsAddr;                                       // lds_addr (bytes)
  g0[2] = (uint32_t)ga;                                  // global_addr[31:0]
  g0[3] = (uint32_t)((ga >> 32) & 0x01FFFFFFu) | (2u << 30); // addr[56:32] | type=2
  i32x8v g1;
  g1[0] = (int)(1u << 16);                               // data_size=1 -> 2 bytes
  g1[1] = (int)((tensorW & 0xFFFFu) << 16);              // tensor_dim0[15:0]
  g1[2] = (int)((tensorW >> 16) | ((tensorH & 0xFFFFu) << 16)); // dim0 hi | dim1 lo
  g1[3] = (int)((tensorH >> 16) | (32u << 16));          // dim1 hi | tile_dim0=32
  g1[4] = (int)(rows & 0xFFFFu);                         // tile_dim1 (tile_dim2=0)
  g1[5] = (int)strideElems;                              // tensor_dim0_stride[31:0]
  g1[6] = 0;                                             // stride hi | dim1_stride lo
  g1[7] = 0;
  i32x4v g2 = {0, 0, 0, 0};
  i32x4v g3 = {0, 0, 0, 0};
#if __clang_major__ >= 23
  i32x8v g4 = {0, 0, 0, 0, 0, 0, 0, 0};
  __builtin_amdgcn_tensor_load_to_lds(g0, g1, g2, g3, g4, 0);
#else
  __builtin_amdgcn_tensor_load_to_lds(g0, g1, g2, g3, 0);
#endif
}
#endif  // HAVE_TDM

// ------------------------------------------------- WMMA bf16 GEMM: C = A*B^T
// A:[M,K] bf16 row-major, B:[N,K] bf16 row-major, C:[M,N] f32 (+ col biases).
// Block tile 64(M) x 128(N), 8 waves in a 2x4 grid, each wave a 2x2 set of
// 16x16 accumulators. Panels staged in LDS (TDM double-buffered when
// available), K-step 32.
// 16x16x32 bf16 fragment layout (ISA 7.12.2):
//   lanes 0-15  : row/col = lane,     K = {0..7, 16..23}
//   lanes 16-31 : row/col = lane-16,  K = {8..15, 24..31}
__device__ __forceinline__ v16bf load_frag_lds(const bf16_t* __restrict__ base,
                                               int row0, int lane) {
  int m  = lane & 15;
  int kh = (lane >> 4) << 3;  // 0 or 8
  const bf16_t* p = base + (row0 + m) * 32 + kh;
  union { v16bf v; uint4 q[2]; } u;
  u.q[0] = *reinterpret_cast<const uint4*>(p);       // ds_load_b128
  u.q[1] = *reinterpret_cast<const uint4*>(p + 16);
  return u.v;
}

__device__ __forceinline__ void mac_step(const bf16_t* __restrict__ Ab,
                                         const bf16_t* __restrict__ Bb,
                                         int wm, int wn, int lane,
                                         v8f& a00, v8f& a01, v8f& a10, v8f& a11) {
  v16bf fa0 = load_frag_lds(Ab, wm * 32,      lane);
  v16bf fa1 = load_frag_lds(Ab, wm * 32 + 16, lane);
  v16bf fb0 = load_frag_lds(Bb, wn * 32,      lane);
  v16bf fb1 = load_frag_lds(Bb, wn * 32 + 16, lane);
  a00 = __builtin_amdgcn_wmma_f32_16x16x32_bf16(false, fa0, false, fb0, (short)0, a00, false, false);
  a01 = __builtin_amdgcn_wmma_f32_16x16x32_bf16(false, fa0, false, fb1, (short)0, a01, false, false);
  a10 = __builtin_amdgcn_wmma_f32_16x16x32_bf16(false, fa1, false, fb0, (short)0, a10, false, false);
  a11 = __builtin_amdgcn_wmma_f32_16x16x32_bf16(false, fa1, false, fb1, (short)0, a11, false, false);
}

__device__ __forceinline__ void store_tile(float* __restrict__ C, int N,
                                           int r0, int c0, v8f acc,
                                           const float* __restrict__ b0,
                                           const float* __restrict__ b1, int lane) {
  int n  = c0 + (lane & 15);
  int mb = r0 + ((lane >> 4) << 3);
  float bv = 0.f;
  if (b0) bv += b0[n];
  if (b1) bv += b1[n];
#pragma unroll
  for (int r = 0; r < 8; ++r)
    C[(size_t)(mb + r) * N + n] = acc[r] + bv;
}

__global__ void __launch_bounds__(256)
k_wmma_gemm_abt(float* __restrict__ C,
                const bf16_t* __restrict__ A,
                const bf16_t* __restrict__ B,
                const float* __restrict__ bias0,
                const float* __restrict__ bias1,
                int M, int N, int K) {
  __shared__ __align__(16) bf16_t As[2][64 * 32];    //  8 KB
  __shared__ __align__(16) bf16_t Bs[2][128 * 32];   // 16 KB
  const int lane = threadIdx.x & 31;
  const int wave = threadIdx.x >> 5;
  const int wm = wave >> 2;               // 0..1  (32-row strip)
  const int wn = wave & 3;                // 0..3  (32-col strip)
  const int row0 = blockIdx.y * 64;
  const int col0 = blockIdx.x * 128;
  const int nK = K >> 5;

  v8f a00 = {}, a01 = {}, a10 = {}, a11 = {};

#if HAVE_TDM
  if (threadIdx.x < 32) {                 // wave 0 drives the TDM pipeline
    tdm_load_2d(lds_off(&As[0][0]), A + (size_t)row0 * K, 64, K, M - row0, K);
    tdm_load_2d(lds_off(&Bs[0][0]), B + (size_t)col0 * K, 128, K, N - col0, K);
  }
  for (int i = 0; i < nK; ++i) {
    const int cur = i & 1;
    if (threadIdx.x < 32) {
      if (i + 1 < nK) {                   // prefetch next K-slab into alt buffer
        const int k1 = (i + 1) << 5;
        tdm_load_2d(lds_off(&As[cur ^ 1][0]), A + (size_t)row0 * K + k1,
                    64, K - k1, M - row0, K);
        tdm_load_2d(lds_off(&Bs[cur ^ 1][0]), B + (size_t)col0 * K + k1,
                    128, K - k1, N - col0, K);
        __builtin_amdgcn_s_wait_tensorcnt(2);   // current A/B pair complete
      } else {
        __builtin_amdgcn_s_wait_tensorcnt(0);
      }
    }
    __syncthreads();                      // LDS tile visible to all waves
    mac_step(&As[cur][0], &Bs[cur][0], wm, wn, lane, a00, a01, a10, a11);
    __syncthreads();                      // done reading before DMA overwrites
  }
#else
  for (int i = 0; i < nK; ++i) {
    const int k0 = i << 5;
    {   // cooperative staging: A 256 x uint4, B 512 x uint4
      int r = threadIdx.x >> 2, s = threadIdx.x & 3;
      *reinterpret_cast<uint4*>(&As[0][r * 32 + s * 8]) =
          *reinterpret_cast<const uint4*>(A + (size_t)(row0 + r) * K + k0 + s * 8);
      for (int v = threadIdx.x; v < 512; v += 256) {
        int r2 = v >> 2, s2 = v & 3;
        *reinterpret_cast<uint4*>(&Bs[0][r2 * 32 + s2 * 8]) =
            *reinterpret_cast<const uint4*>(B + (size_t)(col0 + r2) * K + k0 + s2 * 8);
      }
    }
    __syncthreads();
    mac_step(&As[0][0], &Bs[0][0], wm, wn, lane, a00, a01, a10, a11);
    __syncthreads();
  }
#endif

  store_tile(C, N, row0 + wm * 32,      col0 + wn * 32,      a00, bias0, bias1, lane);
  store_tile(C, N, row0 + wm * 32,      col0 + wn * 32 + 16, a01, bias0, bias1, lane);
  store_tile(C, N, row0 + wm * 32 + 16, col0 + wn * 32,      a10, bias0, bias1, lane);
  store_tile(C, N, row0 + wm * 32 + 16, col0 + wn * 32 + 16, a11, bias0, bias1, lane);
}

// -------------------------------------------- bottom LSTM: persistent scan
// 16 WGs x 64 threads = 1024 threads; thread owns hidden unit j and all 4 of
// its gate rows, so only the 4KB h vector (L2-resident) crosses WGs per step.
__global__ void __launch_bounds__(TPB_SCAN)
k_bot_scan(const float* __restrict__ XG,        // [SEQ, 4H]
           const bf16_t* __restrict__ Whhbf,    // [4H, H] bf16
           float* __restrict__ Y,               // [SEQ, H]
           bf16_t* __restrict__ Ybf,            // [SEQ, H]
           float* __restrict__ hbuf,            // [H], pre-zeroed
           int* cnt, int* gen) {
  __shared__ float hsh[HID];
  const int j = blockIdx.x * TPB_SCAN + threadIdx.x;
  const uint32_t* W32 = reinterpret_cast<const uint32_t*>(Whhbf);
  const uint32_t* wi = W32 + (size_t)j            * (HID / 2);
  const uint32_t* wf = W32 + (size_t)(HID   + j)  * (HID / 2);
  const uint32_t* wg = W32 + (size_t)(2*HID + j)  * (HID / 2);
  const uint32_t* wo = W32 + (size_t)(3*HID + j)  * (HID / 2);

  float c = 0.f;
  for (int t = 0; t < SEQ; ++t) {
    for (int i = threadIdx.x; i < HID; i += TPB_SCAN)
      hsh[i] = __hip_atomic_load(&hbuf[i], __ATOMIC_RELAXED, __HIP_MEMORY_SCOPE_AGENT);
    __syncthreads();

    const float* xg = XG + (size_t)t * G4;
    float gi = xg[j], gf = xg[HID + j], gg = xg[2*HID + j], go = xg[3*HID + j];
#pragma unroll 4
    for (int k2 = 0; k2 < HID / 2; ++k2) {
      float2 h2 = *reinterpret_cast<const float2*>(&hsh[2 * k2]);
      float2 a;
      a = bf2f(wi[k2]); gi += a.x * h2.x + a.y * h2.y;
      a = bf2f(wf[k2]); gf += a.x * h2.x + a.y * h2.y;
      a = bf2f(wg[k2]); gg += a.x * h2.x + a.y * h2.y;
      a = bf2f(wo[k2]); go += a.x * h2.x + a.y * h2.y;
    }
    c = sigm(gf) * c + sigm(gi) * tanhf(gg);
    float h = sigm(go) * tanhf(c);

    hbuf[j] = h;
    Y  [(size_t)t * HID + j] = h;
    Ybf[(size_t)t * HID + j] = (bf16_t)h;
    grid_sync(cnt, gen, NWG_SCAN);
  }
}

// ----------------------------------------- gather up_x rows and cast to bf16
__global__ void k_gather_upx(bf16_t* __restrict__ upx, const float* __restrict__ Y) {
  int i = blockIdx.x;               // 0..63
  int t = 63 + 64 * i;              // linspace(63, 4095, 64)
  for (int k = threadIdx.x; k < HID; k += blockDim.x)
    upx[(size_t)i * HID + k] = (bf16_t)Y[(size_t)t * HID + k];
}

// ------------------------------------------------ top LSTM: one WG, 64 steps
__global__ void __launch_bounds__(1024)
k_top_scan(const float* __restrict__ XGt,       // [UPL, 4H]
           const bf16_t* __restrict__ Whhbf,    // [4H, H] bf16
           float* __restrict__ ctx) {           // [H] final h
  __shared__ float hsh[HID];
  const int j = threadIdx.x;
  const uint32_t* W32 = reinterpret_cast<const uint32_t*>(Whhbf);
  const uint32_t* wi = W32 + (size_t)j            * (HID / 2);
  const uint32_t* wf = W32 + (size_t)(HID   + j)  * (HID / 2);
  const uint32_t* wg = W32 + (size_t)(2*HID + j)  * (HID / 2);
  const uint32_t* wo = W32 + (size_t)(3*HID + j)  * (HID / 2);

  float c = 0.f, h = 0.f;
  hsh[j] = 0.f;
  __syncthreads();
  for (int t = 0; t < UPL; ++t) {
    const float* xg = XGt + (size_t)t * G4;
    float gi = xg[j], gf = xg[HID + j], gg = xg[2*HID + j], go = xg[3*HID + j];
#pragma unroll 4
    for (int k2 = 0; k2 < HID / 2; ++k2) {
      float2 h2 = *reinterpret_cast<const float2*>(&hsh[2 * k2]);
      float2 a;
      a = bf2f(wi[k2]); gi += a.x * h2.x + a.y * h2.y;
      a = bf2f(wf[k2]); gf += a.x * h2.x + a.y * h2.y;
      a = bf2f(wg[k2]); gg += a.x * h2.x + a.y * h2.y;
      a = bf2f(wo[k2]); go += a.x * h2.x + a.y * h2.y;
    }
    c = sigm(gf) * c + sigm(gi) * tanhf(gg);
    h = sigm(go) * tanhf(c);
    __syncthreads();
    hsh[j] = h;
    __syncthreads();
  }
  ctx[j] = h;
}

// ------------- cterm[n] = cw_w[n,:]@ctx + cw_b[n] + att_bias[n] + hw_b[n]
__global__ void k_cterm(float* __restrict__ cterm, const float* __restrict__ cw_w,
                        const float* __restrict__ cw_b, const float* __restrict__ att_bias,
                        const float* __restrict__ hw_b, const float* __restrict__ ctx) {
  int n = blockIdx.x * blockDim.x + threadIdx.x;
  if (n >= HID) return;
  const float* w = cw_w + (size_t)n * HID;
  float s = cw_b[n] + att_bias[n] + hw_b[n];
  for (int k = 0; k < HID; ++k) s += w[k] * ctx[k];
  cterm[n] = s;
}

// -------------------- e[t] = v2s_w . tanh(Mpre[t,:]) + v2s_b  (block reduce)
__global__ void __launch_bounds__(256)
k_escore(float* __restrict__ e, const float* __restrict__ Mpre,
         const float* __restrict__ v2s_w, const float* __restrict__ v2s_b) {
  __shared__ float red[256];
  const int t = blockIdx.x;
  const float* row = Mpre + (size_t)t * HID;
  float s = 0.f;
  for (int k = threadIdx.x; k < HID; k += 256)
    s += tanhf(row[k]) * v2s_w[k];
  red[threadIdx.x] = s;
  __syncthreads();
  for (int off = 128; off > 0; off >>= 1) {
    if (threadIdx.x < off) red[threadIdx.x] += red[threadIdx.x + off];
    __syncthreads();
  }
  if (threadIdx.x == 0) e[t] = red[0] + v2s_b[0];
}

// ------------------------------------------- p[t] = exp(e[t] - max(e)), 1 WG
__global__ void __launch_bounds__(1024)
k_softmax_p(float* __restrict__ p, const float* __restrict__ e) {
  __shared__ float red[1024];
  float m = -3.402823466e+38f;
  for (int i = threadIdx.x; i < SEQ; i += 1024) m = fmaxf(m, e[i]);
  red[threadIdx.x] = m;
  __syncthreads();
  for (int off = 512; off > 0; off >>= 1) {
    if (threadIdx.x < off) red[threadIdx.x] = fmaxf(red[threadIdx.x], red[threadIdx.x + off]);
    __syncthreads();
  }
  float mx = red[0];
  for (int i = threadIdx.x; i < SEQ; i += 1024) p[i] = __expf(e[i] - mx);
}

// ----------------- out[t,c] = cumsum_t(p*y)[t,c] / cumsum_t(p)[t], col/thread
__global__ void __launch_bounds__(256)
k_cumout(float* __restrict__ out, const float* __restrict__ Y, const float* __restrict__ p) {
  const int c = blockIdx.x * 256 + threadIdx.x;   // 0..1023
  float num = 0.f, den = 0.f;
  for (int t = 0; t < SEQ; ++t) {
    float pt = p[t];
    num += pt * Y[(size_t)t * HID + c];
    den += pt;
    out[(size_t)t * HID + c] = num / den;
  }
}

// =========================================================================
extern "C" void kernel_launch(void* const* d_in, const int* in_sizes, int n_in,
                              void* d_out, int out_size, void* d_ws, size_t ws_size,
                              hipStream_t stream) {
  (void)in_sizes; (void)n_in; (void)out_size; (void)ws_size;
  const float* x        = (const float*)d_in[0];
  const float* bot_Wih  = (const float*)d_in[1];
  const float* bot_Whh  = (const float*)d_in[2];
  const float* bot_bih  = (const float*)d_in[3];
  const float* bot_bhh  = (const float*)d_in[4];
  const float* top_Wih  = (const float*)d_in[5];
  const float* top_Whh  = (const float*)d_in[6];
  const float* top_bih  = (const float*)d_in[7];
  const float* top_bhh  = (const float*)d_in[8];
  const float* hw_w     = (const float*)d_in[9];
  const float* hw_b     = (const float*)d_in[10];
  const float* cw_w     = (const float*)d_in[11];
  const float* cw_b     = (const float*)d_in[12];
  const float* att_bias = (const float*)d_in[13];
  const float* v2s_w    = (const float*)d_in[14];
  const float* v2s_b    = (const float*)d_in[15];
  float* out = (float*)d_out;

  // ---- workspace carve-up (~125 MB, 256B aligned)
  char* w = (char*)d_ws;
  auto take = [&](size_t b) { char* r = w; w += (b + 255) & ~(size_t)255; return r; };
  float*  XG     = (float*) take((size_t)SEQ * G4  * 4);  // 64 MB; reused as XGtop
  float*  Y      = (float*) take((size_t)SEQ * HID * 4);  // 16 MB
  bf16_t* Ybf    = (bf16_t*)take((size_t)SEQ * HID * 2);  //  8 MB
  bf16_t* Whhbf  = (bf16_t*)take((size_t)G4  * HID * 2);  //  8 MB; reused for top_Whh
  bf16_t* tWihbf = (bf16_t*)take((size_t)G4  * HID * 2);  //  8 MB
  bf16_t* hwbf   = (bf16_t*)take((size_t)HID * HID * 2);  //  2 MB
  char*   scr    =          take((size_t)SEQ * HID * 4);  // 16 MB (xbf+Wihbf, then Mpre)
  bf16_t* xbf    = (bf16_t*)scr;
  bf16_t* Wihbf  = (bf16_t*)(scr + (size_t)SEQ * INPD * 2);
  float*  Mpre   = (float*) scr;                          // alias: live after scan
  bf16_t* upxbf  = (bf16_t*)take((size_t)UPL * HID * 2);
  float*  hbuf   = (float*) take(HID * 4);
  float*  ctx    = (float*) take(HID * 4);
  float*  cterm  = (float*) take(HID * 4);
  float*  ebuf   = (float*) take(SEQ * 4);
  float*  pbuf   = (float*) take(SEQ * 4);
  int*    sync   = (int*)   take(256);

  // deterministic per-call state init (ws is poisoned once, never re-poisoned)
  hipMemsetAsync(hbuf, 0, HID * 4, stream);
  hipMemsetAsync(sync, 0, 256, stream);

  const int nBig = SEQ * INPD;                  // 4M elems
  // ---- bf16 staging
  k_cvt_bf16<<<(nBig + 255) / 256, 256, 0, stream>>>(xbf,    x,       nBig);
  k_cvt_bf16<<<(nBig + 255) / 256, 256, 0, stream>>>(Wihbf,  bot_Wih, nBig);
  k_cvt_bf16<<<(nBig + 255) / 256, 256, 0, stream>>>(Whhbf,  bot_Whh, nBig);
  k_cvt_bf16<<<(nBig + 255) / 256, 256, 0, stream>>>(tWihbf, top_Wih, nBig);
  k_cvt_bf16<<<(HID * HID + 255) / 256, 256, 0, stream>>>(hwbf, hw_w, HID * HID);

  // ---- bottom gate pre-activations: XG = x @ Wih^T + bih + bhh  (WMMA+TDM)
  k_wmma_gemm_abt<<<dim3(G4 / 128, SEQ / 64), 256, 0, stream>>>(
      XG, xbf, Wihbf, bot_bih, bot_bhh, SEQ, G4, INPD);

  // ---- bottom recurrent scan (persistent, grid-synced)
  k_bot_scan<<<NWG_SCAN, TPB_SCAN, 0, stream>>>(XG, Whhbf, Y, Ybf, hbuf,
                                                &sync[0], &sync[1]);

  // ---- top LSTM over 64 sampled rows
  k_gather_upx<<<UPL, 256, 0, stream>>>(upxbf, Y);
  k_wmma_gemm_abt<<<dim3(G4 / 128, UPL / 64), 256, 0, stream>>>(
      XG /*reuse as XGtop*/, upxbf, tWihbf, top_bih, top_bhh, UPL, G4, HID);
  k_cvt_bf16<<<(nBig + 255) / 256, 256, 0, stream>>>(Whhbf /*reuse*/, top_Whh, nBig);
  k_top_scan<<<1, 1024, 0, stream>>>(XG, Whhbf, ctx);

  // ---- attention: Mpre = y @ hw_w^T + (hw_b + ctx@cw_w^T + cw_b + att_bias)
  k_cterm<<<4, 256, 0, stream>>>(cterm, cw_w, cw_b, att_bias, hw_b, ctx);
  k_wmma_gemm_abt<<<dim3(HID / 128, SEQ / 64), 256, 0, stream>>>(
      Mpre, Ybf, hwbf, cterm, nullptr, SEQ, HID, HID);

  // ---- scores, softmax numerators, running weighted mean
  k_escore<<<SEQ, 256, 0, stream>>>(ebuf, Mpre, v2s_w, v2s_b);
  k_softmax_p<<<1, 1024, 0, stream>>>(pbuf, ebuf);
  k_cumout<<<HID / 256, 256, 0, stream>>>(out, Y, pbuf);
}
